// LSTM_48498770707269
// MI455X (gfx1250) — matrix operand order repaired and checked by reading, not compiled, not run on hardware
//
#include <hip/hip_runtime.h>
#include <math.h>
#include <stdint.h>

// ---------------------------------------------------------------------------
// LSTM on MI455X (gfx1250): bf16 WMMA recurrent GEMMs, fused cell update,
// async global->LDS staging (ASYNCcnt) with LDS double buffering.
//   H = I = 1024, O = 512, B = 64, SEQ = 512.
// ---------------------------------------------------------------------------

typedef __bf16 bf16_t;
typedef __attribute__((ext_vector_type(16))) __bf16 v16bf;
typedef __attribute__((ext_vector_type(8)))  __bf16 v8bf;
typedef __attribute__((ext_vector_type(8)))  float  v8f;
typedef __attribute__((ext_vector_type(4)))  int    v4i;

typedef __attribute__((address_space(1))) v4i gv4i;   // global int4
typedef __attribute__((address_space(3))) v4i lv4i;   // LDS int4

#define HID  1024
#define INP  1024
#define OUTD 512
#define BATCH 64
#define SEQ  512
#define KTOT 2048   // [h | x]

#if __has_builtin(__builtin_amdgcn_global_load_async_to_lds_b128) && \
    __has_builtin(__builtin_amdgcn_s_wait_asynccnt)
#define HAS_ASYNC_LDS 1
#else
#define HAS_ASYNC_LDS 0
#endif

// stage 16 bytes from global to LDS (async DMA if available, else via VGPRs)
__device__ __forceinline__ void stage16(const bf16_t* g, bf16_t* l)
{
#if HAS_ASYNC_LDS
    __builtin_amdgcn_global_load_async_to_lds_b128(
        (gv4i*)(uintptr_t)g,
        (lv4i*)(unsigned)(uintptr_t)l,
        0, 0);
#else
    *(v8bf*)l = *(const v8bf*)g;
#endif
}

__device__ __forceinline__ void stage_fence()
{
#if HAS_ASYNC_LDS
    __builtin_amdgcn_s_wait_asynccnt(0);
#endif
    __syncthreads();
}

// ---------------------------------------------------------------------------
// One-time packing kernels
// ---------------------------------------------------------------------------

// Wcat[g][n][k] : k<1024 -> W_h[g][n][k], else W_x[g][n][k-1024]   (bf16)
__global__ __launch_bounds__(256) void pack_weights(
    const float* __restrict__ whi, const float* __restrict__ whf,
    const float* __restrict__ who, const float* __restrict__ whg,
    const float* __restrict__ wxi, const float* __restrict__ wxf,
    const float* __restrict__ wxo, const float* __restrict__ wxg,
    bf16_t* __restrict__ Wcat)
{
    const float* wh[4] = {whi, whf, who, whg};
    const float* wx[4] = {wxi, wxf, wxo, wxg};
    unsigned idx = blockIdx.x * 256u + threadIdx.x;   // < 4*1024*2048 = 2^23
    unsigned g = idx >> 21;
    unsigned n = (idx >> 11) & 1023u;
    unsigned k = idx & 2047u;
    float v = (k < 1024u) ? wh[g][n * 1024u + k] : wx[g][n * 1024u + (k - 1024u)];
    Wcat[idx] = (bf16_t)v;
}

__global__ __launch_bounds__(256) void pack_wout(
    const float* __restrict__ w, bf16_t* __restrict__ Wout)
{
    unsigned idx = blockIdx.x * 256u + threadIdx.x;   // < 512*1024
    Wout[idx] = (bf16_t)w[idx];
}

__global__ __launch_bounds__(256) void convert_x(
    const float* __restrict__ X, bf16_t* __restrict__ Xbf, unsigned n)
{
    for (unsigned i = blockIdx.x * 256u + threadIdx.x; i < n; i += gridDim.x * 256u)
        Xbf[i] = (bf16_t)X[i];
}

// init h (bf16, broadcast of h0), c (fp32, broadcast of c0), combined biases
__global__ __launch_bounds__(256) void init_state(
    const float* __restrict__ h0, const float* __restrict__ c0,
    const float* __restrict__ bhi, const float* __restrict__ bxi,
    const float* __restrict__ bhf, const float* __restrict__ bxf,
    const float* __restrict__ bho, const float* __restrict__ bxo,
    const float* __restrict__ bhg, const float* __restrict__ bxg,
    bf16_t* __restrict__ hbuf, float* __restrict__ cbuf,
    float* __restrict__ bias)
{
    unsigned idx = blockIdx.x * 256u + threadIdx.x;   // < 65536
    unsigned n = idx & 1023u;
    hbuf[idx] = (bf16_t)h0[n];
    cbuf[idx] = c0[n];
    if (idx < 4096u) {
        const float* bh[4] = {bhi, bhf, bho, bhg};
        const float* bx[4] = {bxi, bxf, bxo, bxg};
        unsigned g = idx >> 10;
        bias[idx] = bh[g][n] + bx[g][n];
    }
}

// ---------------------------------------------------------------------------
// Per-timestep kernel: gates = [h|x_t] (64x2048) @ Wcat^T -> (64 x 4*1024),
// then fused sigmoid/tanh cell update.
// Grid: 64 blocks (16 hidden units each), 256 threads = 8 waves:
//   wave -> (gate = wave>>1, mt-pair = wave&1); each wave owns 2 M-tiles.
// LDS double-buffered; async DMA staging overlaps next chunk with WMMAs.
// ---------------------------------------------------------------------------
__global__ __launch_bounds__(256) void lstm_step(
    const bf16_t* __restrict__ h_in, bf16_t* __restrict__ h_out,
    float* __restrict__ c, const bf16_t* __restrict__ Xbf,
    const bf16_t* __restrict__ Wcat, const float* __restrict__ bias, int t)
{
    __shared__ alignas(64) bf16_t ldsA[2][64][32];      // [buf][batch][k]   8 KB
    __shared__ alignas(64) bf16_t ldsB[2][4][16][32];   // [buf][gate][n][k] 8 KB
    __shared__ alignas(64) float  ldsG[4][64][16];      // pre-activations  16 KB

    const int tid   = threadIdx.x;
    const int lane  = tid & 31;
    const int wave  = tid >> 5;
    const int g     = wave >> 1;        // gate 0..3  (i,f,o,g)
    const int mtp   = wave & 1;         // M-tile pair
    const int nbase = blockIdx.x * 16;

    const int nl    = lane & 15;
    const int khalf = lane >> 4;

    // per-thread staging decode (constant across chunks)
    const int arow = tid >> 2, ack = tid & 3;                 // A: 64 rows x 4 chunks
    const int bgg = tid >> 6, brem = tid & 63;                // B: 4 gates
    const int bu = brem >> 2, bck = brem & 3;                 // 16 units x 4 chunks
    const bf16_t* bsrc_row = &Wcat[(unsigned)(bgg * HID + nbase + bu) * KTOT + bck * 8];

    auto stage_chunk = [&](int k0, int bi) {
        int kk = k0 + ack * 8;
        const bf16_t* srcA = (kk < HID)
            ? &h_in[arow * HID + kk]
            : &Xbf[((unsigned)(arow * SEQ + t)) * INP + (kk - HID)];
        stage16(srcA, &ldsA[bi][arow][ack * 8]);
        stage16(bsrc_row + k0, &ldsB[bi][bgg][bu][bck * 8]);
    };

    v8f acc[2] = {};                    // 2 M-tiles of 16x16 f32

    stage_chunk(0, 0);
    stage_fence();

    for (int i = 0; i < KTOT / 32; ++i) {
        const int cur = i & 1;
        if (i + 1 < KTOT / 32) stage_chunk((i + 1) * 32, cur ^ 1);

        // B fragment (32x16): lane holds column N=nl, K = 16*khalf + e (contig 32B)
        v16bf bfrag = *(const v16bf*)&ldsB[cur][g][nl][khalf * 16];

        #pragma unroll
        for (int j = 0; j < 2; ++j) {
            int mt = mtp * 2 + j;
            // A fragment (16x32): lane holds row M=nl; K = 16*khalf + e (contig 32B)
            v16bf afrag = *(const v16bf*)&ldsA[cur][mt * 16 + nl][khalf * 16];
            acc[j] = __builtin_amdgcn_wmma_f32_16x16x32_bf16(
                false, afrag, false, bfrag, (short)0, acc[j], false, false);
        }
        stage_fence();   // next buffer ready; everyone done reading cur
    }

    // spill pre-activations: D layout M = r + 8*khalf (+16*mt), N = nl
    #pragma unroll
    for (int j = 0; j < 2; ++j)
        #pragma unroll
        for (int r = 0; r < 8; ++r)
            ldsG[g][(mtp * 2 + j) * 16 + r + 8 * khalf][nl] = acc[j][r];
    __syncthreads();

    // fused cell update: 64x16 = 1024 elements / 256 threads
    for (int e = tid; e < 64 * 16; e += 256) {
        int bm = e >> 4;
        int ln = e & 15;
        int n  = nbase + ln;
        float iv = ldsG[0][bm][ln] + bias[0 * HID + n];
        float fv = ldsG[1][bm][ln] + bias[1 * HID + n];
        float ov = ldsG[2][bm][ln] + bias[2 * HID + n];
        float gv = ldsG[3][bm][ln] + bias[3 * HID + n];
        iv = 1.f / (1.f + __expf(-iv));
        fv = 1.f / (1.f + __expf(-fv));
        ov = 1.f / (1.f + __expf(-ov));
        gv = tanhf(gv);
        float cn = fv * c[bm * HID + n] + iv * gv;
        c[bm * HID + n] = cn;
        h_out[bm * HID + n] = (bf16_t)(ov * tanhf(cn));
    }
}

// ---------------------------------------------------------------------------
// Final projection: out (64x512) = h (64x1024 bf16) @ w^T (bf16). 16 blocks
// of 32 output columns; 8 waves -> (mt = wave>>1, nt = wave&1).
// ---------------------------------------------------------------------------
__global__ __launch_bounds__(256) void out_proj(
    const bf16_t* __restrict__ hfin, const bf16_t* __restrict__ Wout,
    float* __restrict__ out)
{
    __shared__ alignas(64) bf16_t ldsA[2][64][32];
    __shared__ alignas(64) bf16_t ldsB[2][32][32];

    const int tid   = threadIdx.x;
    const int lane  = tid & 31;
    const int wave  = tid >> 5;
    const int mt    = wave >> 1;
    const int nt    = wave & 1;
    const int nbase = blockIdx.x * 32;
    const int nl    = lane & 15;
    const int khalf = lane >> 4;

    const int arow = tid >> 2, ack = tid & 3;
    const int bu = (tid & 127) >> 2, bck = tid & 3;
    const bool bload = tid < 128;

    auto stage_chunk = [&](int k0, int bi) {
        stage16(&hfin[arow * HID + k0 + ack * 8], &ldsA[bi][arow][ack * 8]);
        if (bload)
            stage16(&Wout[(unsigned)(nbase + bu) * HID + k0 + bck * 8],
                    &ldsB[bi][bu][bck * 8]);
    };

    v8f acc = {};

    stage_chunk(0, 0);
    stage_fence();

    for (int i = 0; i < HID / 32; ++i) {
        const int cur = i & 1;
        if (i + 1 < HID / 32) stage_chunk((i + 1) * 32, cur ^ 1);

        v16bf bfrag = *(const v16bf*)&ldsB[cur][nt * 16 + nl][khalf * 16];
        v16bf afrag = *(const v16bf*)&ldsA[cur][mt * 16 + nl][khalf * 16];
        acc = __builtin_amdgcn_wmma_f32_16x16x32_bf16(
            false, afrag, false, bfrag, (short)0, acc, false, false);

        stage_fence();
    }

    #pragma unroll
    for (int r = 0; r < 8; ++r) {
        int m = mt * 16 + r + 8 * khalf;
        out[m * OUTD + nbase + nt * 16 + nl] = acc[r];
    }
}

// ---------------------------------------------------------------------------
// Host-side orchestration (graph-capture safe: only launches on `stream`)
// ---------------------------------------------------------------------------
extern "C" void kernel_launch(void* const* d_in, const int* in_sizes, int n_in,
                              void* d_out, int out_size, void* d_ws, size_t ws_size,
                              hipStream_t stream)
{
    (void)in_sizes; (void)n_in; (void)out_size; (void)ws_size;

    const float* X    = (const float*)d_in[0];
    const float* c0   = (const float*)d_in[1];
    const float* h0   = (const float*)d_in[2];
    const float* w_hi = (const float*)d_in[3];
    const float* w_xi = (const float*)d_in[4];
    const float* b_hi = (const float*)d_in[5];
    const float* b_xi = (const float*)d_in[6];
    const float* w_hf = (const float*)d_in[7];
    const float* w_xf = (const float*)d_in[8];
    const float* b_hf = (const float*)d_in[9];
    const float* b_xf = (const float*)d_in[10];
    const float* w_ho = (const float*)d_in[11];
    const float* w_xo = (const float*)d_in[12];
    const float* b_ho = (const float*)d_in[13];
    const float* b_xo = (const float*)d_in[14];
    const float* w_hg = (const float*)d_in[15];
    const float* w_xg = (const float*)d_in[16];
    const float* b_hg = (const float*)d_in[17];
    const float* b_xg = (const float*)d_in[18];
    const float* w    = (const float*)d_in[19];

    char* ws = (char*)d_ws;
    size_t off = 0;
    auto carve = [&](size_t bytes) -> void* {
        void* p = ws + off;
        off += (bytes + 255) & ~(size_t)255;
        return p;
    };

    bf16_t* Wcat = (bf16_t*)carve((size_t)4 * HID * KTOT * sizeof(bf16_t)); // 16 MB
    bf16_t* Wout = (bf16_t*)carve((size_t)OUTD * HID * sizeof(bf16_t));     //  1 MB
    bf16_t* Xbf  = (bf16_t*)carve((size_t)BATCH * SEQ * INP * sizeof(bf16_t)); // 64 MB
    float*  bias = (float*)carve((size_t)4 * HID * sizeof(float));
    float*  cbuf = (float*)carve((size_t)BATCH * HID * sizeof(float));
    bf16_t* hb0  = (bf16_t*)carve((size_t)BATCH * HID * sizeof(bf16_t));
    bf16_t* hb1  = (bf16_t*)carve((size_t)BATCH * HID * sizeof(bf16_t));

    // one-time packing / conversion
    pack_weights<<<(4 * HID * KTOT) / 256, 256, 0, stream>>>(
        w_hi, w_hf, w_ho, w_hg, w_xi, w_xf, w_xo, w_xg, Wcat);
    pack_wout<<<(OUTD * HID) / 256, 256, 0, stream>>>(w, Wout);
    convert_x<<<4096, 256, 0, stream>>>(X, Xbf, (unsigned)BATCH * SEQ * INP);
    init_state<<<(BATCH * HID) / 256, 256, 0, stream>>>(
        h0, c0, b_hi, b_xi, b_hf, b_xf, b_ho, b_xo, b_hg, b_xg, hb0, cbuf, bias);

    // sequential recurrence, double-buffered h
    for (int t = 0; t < SEQ; ++t) {
        bf16_t* hin  = (t & 1) ? hb1 : hb0;
        bf16_t* hout = (t & 1) ? hb0 : hb1;
        lstm_step<<<HID / 16, 256, 0, stream>>>(hin, hout, cbuf, Xbf, Wcat, bias, t);
    }
    // after SEQ=512 steps the final h is in hb0
    out_proj<<<OUTD / 32, 256, 0, stream>>>(hb0, Wout, (float*)d_out);
}
